// Switch_15161234555447
// MI455X (gfx1250) — matrix operand order repaired and checked by reading, not compiled, not run on hardware
//
#include <hip/hip_runtime.h>
#include <math.h>

// ---------------------------------------------------------------------------
// Types for CDNA5 WMMA
// ---------------------------------------------------------------------------
typedef __attribute__((ext_vector_type(16))) __bf16 v16bf;
typedef __attribute__((ext_vector_type(8)))  float  v8f;
typedef __attribute__((ext_vector_type(4)))  int    v4i;

struct Frag2x16B { uint4 a, b; };   // 32 bytes == v16bf

// round-to-nearest-even f32 -> bf16 bits
static __device__ __forceinline__ unsigned short f2bf(float f) {
    unsigned int u = __builtin_bit_cast(unsigned int, f);
    unsigned int r = u + 0x7FFFu + ((u >> 16) & 1u);
    return (unsigned short)(r >> 16);
}
static __device__ __forceinline__ float bf2f(unsigned short h) {
    unsigned int u = ((unsigned int)h) << 16;
    return __builtin_bit_cast(float, u);
}

static __device__ __forceinline__ v16bf load_frag(const unsigned short* p, int o1, int o2) {
    Frag2x16B f;
    f.a = *(const uint4*)(p + o1);
    f.b = *(const uint4*)(p + o2);
    return __builtin_bit_cast(v16bf, f);
}

static __device__ __forceinline__ float gelu_f(float v) {
    const float k = 0.7978845608028654f;  // sqrt(2/pi)
    return 0.5f * v * (1.0f + tanhf(k * (v + 0.044715f * v * v * v)));
}

// ---------------------------------------------------------------------------
// Async global->LDS 16-byte copy (CDNA5 GLOBAL_LOAD_ASYNC_TO_LDS_B128).
// Builtin params (per hipcc diagnostic): (int4 AS1*, int4 AS3*, imm, imm).
// Address-space pointers built via integer casts (LDS generic ptr low 32 bits
// are the DS byte offset per the ISA flat->LDS mapping).
// ---------------------------------------------------------------------------
#if defined(__has_builtin) && __has_builtin(__builtin_amdgcn_global_load_async_to_lds_b128)
#define HAVE_ASYNC_LDS 1
#else
#define HAVE_ASYNC_LDS 0
#endif

static __device__ __forceinline__ void async_copy16(const void* g, void* l) {
#if HAVE_ASYNC_LDS
    __builtin_amdgcn_global_load_async_to_lds_b128(
        (__attribute__((address_space(1))) v4i*)(uintptr_t)g,
        (__attribute__((address_space(3))) v4i*)(unsigned int)(uintptr_t)l,
        0, 0);
#else
    *(uint4*)l = *(const uint4*)g;
#endif
}

static __device__ __forceinline__ void async_wait0() {
#if HAVE_ASYNC_LDS
#if __has_builtin(__builtin_amdgcn_s_wait_asynccnt)
    __builtin_amdgcn_s_wait_asynccnt(0);
#else
    asm volatile("s_wait_asynccnt 0" ::: "memory");
#endif
#endif
}

// ---------------------------------------------------------------------------
// 1) Router: one wave (32 lanes) per token. logits = x[t] @ rw ; gate = max
//    softmax prob = 1/sum(exp(l - lmax)); eidx = first argmax.
// ---------------------------------------------------------------------------
__global__ __launch_bounds__(256) void router_kernel(
    const float* __restrict__ x, const float* __restrict__ rw,
    float* __restrict__ gate, int* __restrict__ eidx, int T, int D)
{
    const int wid  = threadIdx.x >> 5;
    const int lane = threadIdx.x & 31;
    const int t = blockIdx.x * 8 + wid;
    if (t >= T) return;

    float acc[8] = {0.f,0.f,0.f,0.f,0.f,0.f,0.f,0.f};
    const float* xr = x + (size_t)t * D;
    for (int d = lane; d < D; d += 32) {
        const float xv = xr[d];
        const float4 w0 = *(const float4*)&rw[d * 8 + 0];
        const float4 w1 = *(const float4*)&rw[d * 8 + 4];
        acc[0] += xv * w0.x; acc[1] += xv * w0.y;
        acc[2] += xv * w0.z; acc[3] += xv * w0.w;
        acc[4] += xv * w1.x; acc[5] += xv * w1.y;
        acc[6] += xv * w1.z; acc[7] += xv * w1.w;
    }
#pragma unroll
    for (int off = 16; off > 0; off >>= 1) {
#pragma unroll
        for (int e = 0; e < 8; ++e) acc[e] += __shfl_xor(acc[e], off, 32);
    }
    if (lane == 0) {
        float lmax = acc[0]; int imax = 0;
#pragma unroll
        for (int e = 1; e < 8; ++e)
            if (acc[e] > lmax) { lmax = acc[e]; imax = e; }
        float s = 0.f;
#pragma unroll
        for (int e = 0; e < 8; ++e) s += expf(acc[e] - lmax);
        gate[t] = 1.0f / s;   // == max softmax prob
        eidx[t] = imax;
    }
}

// ---------------------------------------------------------------------------
// 2) Ordered scan: reproduce position_in_expert = cumsum over token order.
//    Single block, 256 threads x 16 tokens. 1-indexed positions; keep p < C.
// ---------------------------------------------------------------------------
__global__ __launch_bounds__(256) void scan_kernel(
    const int* __restrict__ eidx, int* __restrict__ slot,
    int* __restrict__ tok_of, int T, int C)
{
    __shared__ int cnt[256][8];
    const int tid = threadIdx.x;
    const int per = T / 256;          // 16

    for (int i = tid; i < 8 * C; i += 256) tok_of[i] = -1;

    int local[8] = {0,0,0,0,0,0,0,0};
    const int base = tid * per;
    for (int i = 0; i < per; ++i) local[eidx[base + i]]++;
#pragma unroll
    for (int e = 0; e < 8; ++e) cnt[tid][e] = local[e];
    __syncthreads();

    if (tid == 0) {
        int run[8] = {0,0,0,0,0,0,0,0};
        for (int th = 0; th < 256; ++th)
#pragma unroll
            for (int e = 0; e < 8; ++e) {
                int v = cnt[th][e]; cnt[th][e] = run[e]; run[e] += v;
            }
    }
    __syncthreads();

    int off[8];
#pragma unroll
    for (int e = 0; e < 8; ++e) off[e] = cnt[tid][e];
    for (int i = 0; i < per; ++i) {
        const int t = base + i;
        const int e = eidx[t];
        const int p = ++off[e];          // 1-indexed position in expert
        if (p < C) {                     // strict: positions 1..C-1 kept
            slot[t] = p;
            tok_of[e * C + p] = t;
        } else {
            slot[t] = -1;
        }
    }
}

// ---------------------------------------------------------------------------
// 3) Gather tokens into dense expert buffers, pre-split into bf16 hi/lo
//    planes Xhi/Xlo [E*C, D] (zeros elsewhere, incl. slot 0).
// ---------------------------------------------------------------------------
__global__ __launch_bounds__(256) void gather_split_kernel(
    const float* __restrict__ x, const int* __restrict__ tok_of,
    unsigned short* __restrict__ Xhi, unsigned short* __restrict__ Xlo, int D)
{
    const int row = blockIdx.x;
    const int t = tok_of[row];
    const size_t base = (size_t)row * D;
    if (t >= 0) {
        const float* src = x + (size_t)t * D;
        for (int d = threadIdx.x; d < D; d += blockDim.x) {
            const float f = src[d];
            const unsigned short hb = f2bf(f);
            Xhi[base + d] = hb;
            Xlo[base + d] = f2bf(f - bf2f(hb));
        }
    } else {
        for (int d = threadIdx.x; d < D; d += blockDim.x) {
            Xhi[base + d] = 0;
            Xlo[base + d] = 0;
        }
    }
}

// ---------------------------------------------------------------------------
// 4) Per-expert GEMM + bias + gelu via split-bf16 WMMA:
//    out[m,n] = gelu( sum_k A[m,k] * W[k,n] + bias[n] )
//    A comes pre-split as bf16 hi/lo planes -> A staging is an async DMA copy.
//    W (fp32 input) is split in-register while staging (co-executes w/ WMMA).
//    Block tile 128x64, K-step 32, 8 waves each computing 32x32.
//    3 WMMAs (hi*hi, hi*lo, lo*hi) per 16x16 C tile per K-step.
//    SPLIT_OUT: write result as bf16 hi/lo planes (feeds next GEMM) instead
//    of fp32.
// ---------------------------------------------------------------------------
#define LDSP 40   // padded row pitch in bf16 elements (80 B) -> bank-safe

template <bool SPLIT_OUT>
__global__ __launch_bounds__(256) void expert_gemm_kernel(
    const unsigned short* __restrict__ Ahi, const unsigned short* __restrict__ Alo,
    const float* __restrict__ W, const float* __restrict__ bias,
    float* __restrict__ Cf32,
    unsigned short* __restrict__ Chi, unsigned short* __restrict__ Clo,
    int Mdim, int Ndim, int Kdim)
{
    const int e = blockIdx.z;
    Ahi  += (size_t)e * Mdim * Kdim;
    Alo  += (size_t)e * Mdim * Kdim;
    W    += (size_t)e * Kdim * Ndim;
    bias += (size_t)e * Ndim;
    const size_t obase = (size_t)e * Mdim * Ndim;

    const int m0 = blockIdx.y * 128;
    const int n0 = blockIdx.x * 64;
    const int tid   = threadIdx.x;
    const int lane  = tid & 31;
    const int wid   = tid >> 5;
    const int waveM = wid & 3;        // 0..3  -> 32-row strip
    const int waveN = wid >> 2;       // 0..1  -> 32-col strip
    const int half  = lane >> 4;      // hi/lo 16-lane group
    const int lm    = lane & 15;

    __shared__ unsigned short sAhi[128 * LDSP];
    __shared__ unsigned short sAlo[128 * LDSP];
    __shared__ unsigned short sBhi[64 * LDSP];
    __shared__ unsigned short sBlo[64 * LDSP];

    v8f acc[2][2];
#pragma unroll
    for (int mi = 0; mi < 2; ++mi)
#pragma unroll
        for (int ni = 0; ni < 2; ++ni) acc[mi][ni] = 0.f;

    for (int k0 = 0; k0 < Kdim; k0 += 32) {
        __syncthreads();
        // ---- stage A tile: async DMA of pre-split bf16 planes -------------
        // 128 rows x 32 bf16 = 128 x 4 x 16B chunks per plane; 2 chunks per
        // thread per plane.
        {
#pragma unroll
            for (int pass = 0; pass < 2; ++pass) {
                const int q   = tid + pass * 256;    // 0..511
                const int row = q >> 2;
                const int c16 = (q & 3) * 8;         // ushort offset in row
                const size_t gsrc = (size_t)(m0 + row) * Kdim + k0 + c16;
                async_copy16(&Ahi[gsrc], &sAhi[row * LDSP + c16]);
                async_copy16(&Alo[gsrc], &sAlo[row * LDSP + c16]);
            }
        }
        // ---- stage B tile: 32 x 64 fp32 -> transposed hi/lo bf16 ----------
        {
            const int kk = tid >> 4;             // 0..15
            const int n4 = (tid & 15) * 4;       // 0..60
#pragma unroll
            for (int p = 0; p < 2; ++p) {
                const int krow = p * 16 + kk;
                const float4 v =
                    *(const float4*)&W[(size_t)(k0 + krow) * Ndim + n0 + n4];
                const float f[4] = {v.x, v.y, v.z, v.w};
#pragma unroll
                for (int j = 0; j < 4; ++j) {
                    const unsigned short hb = f2bf(f[j]);
                    const unsigned short lb = f2bf(f[j] - bf2f(hb));
                    sBhi[(n4 + j) * LDSP + krow] = hb;
                    sBlo[(n4 + j) * LDSP + krow] = lb;
                }
            }
        }
        if (k0 + 32 < Kdim) {   // prefetch next W K-tile -> global_prefetch_b8
            __builtin_prefetch(&W[(size_t)(k0 + 32 + (tid >> 4)) * Ndim + n0], 0, 1);
        }
        async_wait0();          // ASYNCcnt==0: this wave's LDS writes landed
        __syncthreads();

        // ---- fragments (layout per ISA 7.12.2, 16-bit A / B) --------------
        v16bf aH[2], aL[2], bH[2], bL[2];
#pragma unroll
        for (int mi = 0; mi < 2; ++mi) {
            const int r = (waveM * 32 + mi * 16 + lm) * LDSP;
            aH[mi] = load_frag(sAhi, r + half * 8, r + 16 + half * 8);
            aL[mi] = load_frag(sAlo, r + half * 8, r + 16 + half * 8);
        }
#pragma unroll
        for (int ni = 0; ni < 2; ++ni) {
            const int r = (waveN * 32 + ni * 16 + lm) * LDSP;
            bH[ni] = load_frag(sBhi, r + half * 8, r + 16 + half * 8);
            bL[ni] = load_frag(sBlo, r + half * 8, r + 16 + half * 8);
        }
        // ---- 12 WMMAs: hi*hi + hi*lo + lo*hi per 16x16 C tile -------------
#pragma unroll
        for (int mi = 0; mi < 2; ++mi)
#pragma unroll
            for (int ni = 0; ni < 2; ++ni) {
                acc[mi][ni] = __builtin_amdgcn_wmma_f32_16x16x32_bf16(
                    false, aH[mi], false, bH[ni], (short)0, acc[mi][ni], false, false);
                acc[mi][ni] = __builtin_amdgcn_wmma_f32_16x16x32_bf16(
                    false, aH[mi], false, bL[ni], (short)0, acc[mi][ni], false, false);
                acc[mi][ni] = __builtin_amdgcn_wmma_f32_16x16x32_bf16(
                    false, aL[mi], false, bH[ni], (short)0, acc[mi][ni], false, false);
            }
    }

    // ---- epilogue: bias + gelu; C/D layout per ISA (VGPR r, lane halves) ---
#pragma unroll
    for (int mi = 0; mi < 2; ++mi)
#pragma unroll
        for (int ni = 0; ni < 2; ++ni) {
            const int row0 = m0 + waveM * 32 + mi * 16 + half * 8;
            const int col  = n0 + waveN * 32 + ni * 16 + lm;
            const float bc = bias[col];
#pragma unroll
            for (int r = 0; r < 8; ++r) {
                const float g = gelu_f(acc[mi][ni][r] + bc);
                const size_t idx = obase + (size_t)(row0 + r) * Ndim + col;
                if (SPLIT_OUT) {
                    const unsigned short hb = f2bf(g);
                    Chi[idx] = hb;
                    Clo[idx] = f2bf(g - bf2f(hb));
                } else {
                    Cf32[idx] = g;
                }
            }
        }
}

// ---------------------------------------------------------------------------
// 5) Combine back to token order: out[t] = gate[t] * Y[e_t, slot_t] or 0.
// ---------------------------------------------------------------------------
__global__ __launch_bounds__(256) void combine_kernel(
    const float* __restrict__ Y, const float* __restrict__ gate,
    const int* __restrict__ eidx, const int* __restrict__ slot,
    float* __restrict__ out, int D, int C)
{
    const int t = blockIdx.x;
    const int s = slot[t];
    float* dst = out + (size_t)t * D;
    if (s >= 0) {
        const float g = gate[t];
        const float* src = Y + ((size_t)eidx[t] * C + s) * D;
        for (int d = threadIdx.x; d < D; d += blockDim.x) dst[d] = g * src[d];
    } else {
        for (int d = threadIdx.x; d < D; d += blockDim.x) dst[d] = 0.0f;
    }
}

// ---------------------------------------------------------------------------
// Launch
// ---------------------------------------------------------------------------
extern "C" void kernel_launch(void* const* d_in, const int* in_sizes, int n_in,
                              void* d_out, int out_size, void* d_ws, size_t ws_size,
                              hipStream_t stream) {
    (void)in_sizes; (void)n_in; (void)out_size; (void)ws_size;

    const float* x  = (const float*)d_in[0];  // [B,S,D]
    const float* rw = (const float*)d_in[1];  // [D,E]
    const float* w1 = (const float*)d_in[2];  // [E,D,M]
    const float* b1 = (const float*)d_in[3];  // [E,M]
    const float* w2 = (const float*)d_in[4];  // [E,M,D]
    const float* b2 = (const float*)d_in[5];  // [E,D]
    float* out = (float*)d_out;

    const int Bb = 8, Ss = 512, D = 768, M = 3072, E = 8;
    const int T = Bb * Ss;       // 4096
    const int C = T / E;         // 512

    char* ws = (char*)d_ws;
    size_t o = 0;
    auto wsalloc = [&](size_t bytes) -> void* {
        void* p = ws + o;
        o = (o + bytes + 255) & ~(size_t)255;
        return p;
    };
    float* gate  = (float*)wsalloc((size_t)T * 4);
    int*   eidx  = (int*)  wsalloc((size_t)T * 4);
    int*   slot  = (int*)  wsalloc((size_t)T * 4);
    int*   tokof = (int*)  wsalloc((size_t)E * C * 4);
    unsigned short* Xhi = (unsigned short*)wsalloc((size_t)E * C * D * 2); // 6.3 MB
    unsigned short* Xlo = (unsigned short*)wsalloc((size_t)E * C * D * 2); // 6.3 MB
    unsigned short* Hhi = (unsigned short*)wsalloc((size_t)E * C * M * 2); // 25.2 MB
    unsigned short* Hlo = (unsigned short*)wsalloc((size_t)E * C * M * 2); // 25.2 MB
    float* Y            = (float*)wsalloc((size_t)E * C * D * 4);          // 12.6 MB

    router_kernel<<<T / 8, 256, 0, stream>>>(x, rw, gate, eidx, T, D);
    scan_kernel<<<1, 256, 0, stream>>>(eidx, slot, tokof, T, C);
    gather_split_kernel<<<E * C, 256, 0, stream>>>(x, tokof, Xhi, Xlo, D);
    // H = gelu(Xe @ w1 + b1): Mdim=C, Ndim=M, Kdim=D ; H written as hi/lo
    expert_gemm_kernel<true><<<dim3(M / 64, C / 128, E), 256, 0, stream>>>(
        Xhi, Xlo, w1, b1, nullptr, Hhi, Hlo, C, M, D);
    // Y = gelu(H @ w2 + b2): Mdim=C, Ndim=D, Kdim=M ; Y written fp32
    expert_gemm_kernel<false><<<dim3(D / 64, C / 128, E), 256, 0, stream>>>(
        Hhi, Hlo, w2, b2, Y, nullptr, nullptr, C, D, M);
    combine_kernel<<<T, 256, 0, stream>>>(Y, gate, eidx, slot, out, D, C);
}
